// CrysFormerLayer_12841952215475
// MI455X (gfx1250) — compile-verified, matrix operand order checked by reading, not run on hardware
//
#include <hip/hip_runtime.h>
#include <hip/hip_bf16.h>

// ---------------------------------------------------------------------------
// CrysFormer layer for MI455X (gfx1250, wave32).
// GEMM-shaped work -> v_wmma_f32_16x16x32_bf16 (bf16 operands, f32 accum).
// Weights are pre-swizzled into B-fragment order so each fragment is two
// coalesced b128 loads per lane; A fragments are two b128 loads per lane.
// Hot loops are 2-way interleaved (two C accumulators in flight) to fill the
// WMMA->VALU co-exec hazard slots instead of v_nop padding.
// Scatter/softmax pieces use VALU + float atomics (L2-resident gathers).
// ---------------------------------------------------------------------------

#define N_  10000
#define E_  160000
#define C_  64
#define H_  8
#define HC_ 512        // H*C
#define NT_ (N_/16)    // 625 node tiles
#define ET_ (E_/16)    // 10000 edge tiles

typedef __attribute__((ext_vector_type(16))) __bf16 v16bf;
typedef __attribute__((ext_vector_type(8)))  __bf16 v8bf;
typedef __attribute__((ext_vector_type(8)))  float  v8f;

#define WMMA_BF16(A, B, Cacc) \
  __builtin_amdgcn_wmma_f32_16x16x32_bf16(false, (A), false, (B), (short)0, (Cacc), false, false)

// Fragment half-order (linear in the v16bf): for lane-half hf (0/1),
// halves 0..7  hold K = kh+0 .. kh+7        (kh = hf*8)
// halves 8..15 hold K = 16+kh+0 .. 16+kh+7
// (matches ISA 16-bit A/B VGPR layout tables.)

__device__ __forceinline__ v16bf cat8(v8bf lo, v8bf hi) {
  return __builtin_shufflevector(lo, hi, 0, 1, 2, 3, 4, 5, 6, 7,
                                 8, 9, 10, 11, 12, 13, 14, 15);
}

// A fragment (16x32) from row-major bf16, p = &(row0, k0), leading dim ld
// (ld multiple of 8, p 16B aligned). Two b128 loads per lane.
__device__ __forceinline__ v16bf ldA_vec(const __bf16* p, int ld) {
  int lane = threadIdx.x & 31;
  const __bf16* q = p + (lane & 15) * ld + (lane >> 4) * 8;
  v8bf lo = *(const v8bf*)q;
  v8bf hi = *(const v8bf*)(q + 16);
  return cat8(lo, hi);
}

// B fragment from pre-swizzled weight storage: fragment f, lane-contiguous.
__device__ __forceinline__ v16bf ldB_swz(const __bf16* base, int f) {
  const __bf16* p = base + ((size_t)f * 32 + (threadIdx.x & 31)) * 16;
  v8bf lo = *(const v8bf*)p;
  v8bf hi = *(const v8bf*)(p + 8);
  return cat8(lo, hi);
}

// A fragment from row-major f32 (LDS), cast to bf16 (scalar gather, cheap in LDS).
__device__ __forceinline__ v16bf ldA_f32(const float* p, int ld) {
  int lane = threadIdx.x & 31, row = lane & 15, kh = (lane >> 4) * 8;
  v16bf a;
#pragma unroll
  for (int j = 0; j < 8; ++j) {
    a[j]     = (__bf16)p[row * ld + kh + j];
    a[j + 8] = (__bf16)p[row * ld + 16 + kh + j];
  }
  return a;
}

// A fragment of x = concat(u, v, u-v) (16 x 192), k-step base k0.
__device__ __forceinline__ float gated_elem(const float* u, const float* v, int row, int k) {
  if (k < 64)  return u[row * 64 + k];
  if (k < 128) return v[row * 64 + (k - 64)];
  int kk = k - 128;
  return u[row * 64 + kk] - v[row * 64 + kk];
}
__device__ __forceinline__ v16bf ldA_gated(const float* u, const float* v, int k0) {
  int lane = threadIdx.x & 31, row = lane & 15, kh = (lane >> 4) * 8;
  v16bf a;
#pragma unroll
  for (int j = 0; j < 8; ++j) {
    a[j]     = (__bf16)gated_elem(u, v, row, k0 + kh + j);
    a[j + 8] = (__bf16)gated_elem(u, v, row, k0 + 16 + kh + j);
  }
  return a;
}

__device__ __forceinline__ float silu_f(float x) { return x / (1.0f + __expf(-x)); }

__device__ __forceinline__ void atomicMaxFloat(float* addr, float val) {
  if (!(val < 0.0f)) atomicMax((int*)addr, __float_as_int(val));
  else               atomicMin((unsigned int*)addr, __float_as_uint(val));
}

// ---------------------------------------------------------------------------
// Prep kernels
// ---------------------------------------------------------------------------

// Convert + swizzle a row-major f32 weight [KT*32, CT*16] into B-fragment
// order: dst[((kt*CT+ct)*32 + lane)*16 + j] in the fragment half-order above.
__global__ void __launch_bounds__(32)
k_swzB(const float* __restrict__ W, __bf16* __restrict__ dst, int CT, int ncols) {
  int f = blockIdx.x;
  int kt = f / CT, ct = f % CT;
  int lane = threadIdx.x;
  int col = ct * 16 + (lane & 15);
  int kh = (lane >> 4) * 8;
  __bf16* d = dst + ((size_t)f * 32 + lane) * 16;
#pragma unroll
  for (int j = 0; j < 8; ++j) {
    d[j]     = (__bf16)W[(kt * 32 + kh + j) * ncols + col];
    d[j + 8] = (__bf16)W[(kt * 32 + 16 + kh + j) * ncols + col];
  }
}

// f32 -> bf16, 4 elements per thread (vector load, packed store).
__global__ void k_f32_to_bf16_v4(const float* __restrict__ src, __bf16* __restrict__ dst, int n4) {
  int i = blockIdx.x * blockDim.x + threadIdx.x;
  if (i >= n4) return;
  float4 f = ((const float4*)src)[i];
  typedef __attribute__((ext_vector_type(4))) __bf16 v4bf;
  v4bf o; o[0] = (__bf16)f.x; o[1] = (__bf16)f.y; o[2] = (__bf16)f.z; o[3] = (__bf16)f.w;
  ((v4bf*)dst)[i] = o;
}

__global__ void k_fill(float* __restrict__ p, float val, int n) {
  int i = blockIdx.x * blockDim.x + threadIdx.x;
  if (i < n) p[i] = val;
}

// ---------------------------------------------------------------------------
// LayerNorm over C=64: one wave per row, 2 elements per lane.
// ---------------------------------------------------------------------------
__global__ void __launch_bounds__(32)
k_ln(const float* __restrict__ x, const float* __restrict__ g, const float* __restrict__ b,
     float* __restrict__ y, __bf16* __restrict__ ybf) {
  int row = blockIdx.x, lane = threadIdx.x;
  float x0 = x[row * 64 + lane], x1 = x[row * 64 + 32 + lane];
  float s = x0 + x1;
#pragma unroll
  for (int m = 16; m; m >>= 1) s += __shfl_xor(s, m, 32);
  float mu = s * (1.0f / 64.0f);
  float d0 = x0 - mu, d1 = x1 - mu;
  float vs = d0 * d0 + d1 * d1;
#pragma unroll
  for (int m = 16; m; m >>= 1) vs += __shfl_xor(vs, m, 32);
  float inv = rsqrtf(vs * (1.0f / 64.0f) + 1e-5f);
  float y0 = d0 * inv * g[lane] + b[lane];
  float y1 = d1 * inv * g[lane + 32] + b[lane + 32];
  y[row * 64 + lane] = y0;
  y[row * 64 + 32 + lane] = y1;
  ybf[row * 64 + lane] = (__bf16)y0;
  ybf[row * 64 + 32 + lane] = (__bf16)y1;
}

// ---------------------------------------------------------------------------
// q/k/v/skip projections. blockIdx.x = matrix id (0..3), blockIdx.y = node tile.
// Two column tiles in flight per iteration (fills WMMA hazard slots).
// ---------------------------------------------------------------------------
__global__ void __launch_bounds__(32)
k_proj(const __bf16* __restrict__ h_bf,
       const __bf16* Wq, const __bf16* Wk, const __bf16* Wv, const __bf16* Ws,
       const float* bq, const float* bk, const float* bv, const float* bs,
       float* q, float* k, float* v, float* skip) {
  int id = blockIdx.x, nt = blockIdx.y, lane = threadIdx.x;
  const __bf16* W; const float* bias; float* out; int cols;
  if      (id == 0) { W = Wq; bias = bq; out = q;    cols = HC_; }
  else if (id == 1) { W = Wk; bias = bk; out = k;    cols = HC_; }
  else if (id == 2) { W = Wv; bias = bv; out = v;    cols = HC_; }
  else              { W = Ws; bias = bs; out = skip; cols = C_;  }
  int CT = cols / 16;

  v16bf a0 = ldA_vec(h_bf + nt * 16 * 64, 64);
  v16bf a1 = ldA_vec(h_bf + nt * 16 * 64 + 32, 64);
  int hf = lane >> 4, cl = lane & 15;
  for (int ct = 0; ct < CT; ct += 2) {
    v8f c0 = {}, c1 = {};
    c0 = WMMA_BF16(a0, ldB_swz(W, ct), c0);
    c1 = WMMA_BF16(a0, ldB_swz(W, ct + 1), c1);
    c0 = WMMA_BF16(a1, ldB_swz(W, CT + ct), c0);
    c1 = WMMA_BF16(a1, ldB_swz(W, CT + ct + 1), c1);
    int col0 = ct * 16 + cl, col1 = col0 + 16;
    float bb0 = bias[col0], bb1 = bias[col1];
#pragma unroll
    for (int i = 0; i < 8; ++i)
      out[(nt * 16 + i + hf * 8) * cols + col0] = c0[i] + bb0;
#pragma unroll
    for (int i = 0; i < 8; ++i)
      out[(nt * 16 + i + hf * 8) * cols + col1] = c1[i] + bb1;
  }
}

// ---------------------------------------------------------------------------
// Edge pass 1: e = edge_attr @ We (WMMA, per head), alpha = q[dst].(k[src]+e)/8,
// per-dst atomic float max. Two 16-col tiles in flight per head half.
// ---------------------------------------------------------------------------
__global__ void __launch_bounds__(32)
k_edge_alpha(const __bf16* __restrict__ eattr_bf, const __bf16* __restrict__ We_s,
             const int* __restrict__ ei, const float* __restrict__ q,
             const float* __restrict__ kk, float* __restrict__ alpha,
             float* __restrict__ amax) {
  int et = blockIdx.x, lane = threadIdx.x;
  int hf = lane >> 4, cl = lane & 15;
  int srcr[8], dstr[8];
#pragma unroll
  for (int i = 0; i < 8; ++i) {
    int e = et * 16 + i + hf * 8;
    srcr[i] = ei[e];
    dstr[i] = ei[E_ + e];
  }
  v16bf a0 = ldA_vec(eattr_bf + (size_t)et * 16 * 64, 64);
  v16bf a1 = ldA_vec(eattr_bf + (size_t)et * 16 * 64 + 32, 64);
  if (et + 1 < ET_)
    __builtin_prefetch(eattr_bf + (size_t)(et + 1) * 16 * 64 + lane * 32, 0, 3);

  for (int h = 0; h < H_; ++h) {
    float acc[8] = {0, 0, 0, 0, 0, 0, 0, 0};
#pragma unroll
    for (int ct = 0; ct < 4; ct += 2) {
      int cidx = h * 4 + ct;
      v8f c0 = {}, c1 = {};
      c0 = WMMA_BF16(a0, ldB_swz(We_s, cidx), c0);
      c1 = WMMA_BF16(a0, ldB_swz(We_s, cidx + 1), c1);
      c0 = WMMA_BF16(a1, ldB_swz(We_s, 32 + cidx), c0);
      c1 = WMMA_BF16(a1, ldB_swz(We_s, 32 + cidx + 1), c1);
      int col0 = cidx * 16 + cl, col1 = col0 + 16;
#pragma unroll
      for (int i = 0; i < 8; ++i)
        acc[i] += q[dstr[i] * HC_ + col0] * (kk[srcr[i] * HC_ + col0] + c0[i]);
#pragma unroll
      for (int i = 0; i < 8; ++i)
        acc[i] += q[dstr[i] * HC_ + col1] * (kk[srcr[i] * HC_ + col1] + c1[i]);
    }
#pragma unroll
    for (int i = 0; i < 8; ++i) {
      float s = acc[i];
      s += __shfl_xor(s, 1, 32); s += __shfl_xor(s, 2, 32);
      s += __shfl_xor(s, 4, 32); s += __shfl_xor(s, 8, 32);
      if (cl == 0) {
        float al = s * 0.125f;  // / sqrt(C)
        int e = et * 16 + i + hf * 8;
        alpha[e * H_ + h] = al;
        atomicMaxFloat(&amax[dstr[i] * H_ + h], al);
      }
    }
  }
}

// Softmax normalize: ex = exp(alpha - amax[dst]); denom[dst] += ex.
__global__ void k_softmax(const int* __restrict__ ei, float* __restrict__ alpha,
                          const float* __restrict__ amax, float* __restrict__ denom) {
  int idx = blockIdx.x * blockDim.x + threadIdx.x;
  if (idx >= E_ * H_) return;
  int e = idx / H_, h = idx % H_;
  int d = ei[E_ + e];
  float ex = __expf(alpha[idx] - amax[d * H_ + h]);
  alpha[idx] = ex;
  atomicAdd(&denom[d * H_ + h], ex);
}

// ---------------------------------------------------------------------------
// Edge pass 2: recompute e (WMMA), msg = (v[src]+e)*w, head-mean atomicAdd.
// ---------------------------------------------------------------------------
__global__ void __launch_bounds__(32)
k_edge_msg(const __bf16* __restrict__ eattr_bf, const __bf16* __restrict__ We_s,
           const int* __restrict__ ei, const float* __restrict__ vmat,
           const float* __restrict__ ex, const float* __restrict__ denom,
           float* __restrict__ out_acc) {
  int et = blockIdx.x, lane = threadIdx.x;
  int hf = lane >> 4, cl = lane & 15;
  int srcr[8], dstr[8];
#pragma unroll
  for (int i = 0; i < 8; ++i) {
    int e = et * 16 + i + hf * 8;
    srcr[i] = ei[e];
    dstr[i] = ei[E_ + e];
  }
  v16bf a0 = ldA_vec(eattr_bf + (size_t)et * 16 * 64, 64);
  v16bf a1 = ldA_vec(eattr_bf + (size_t)et * 16 * 64 + 32, 64);
  if (et + 1 < ET_)
    __builtin_prefetch(eattr_bf + (size_t)(et + 1) * 16 * 64 + lane * 32, 0, 3);

  float macc[4][8];
#pragma unroll
  for (int ct = 0; ct < 4; ++ct)
#pragma unroll
    for (int i = 0; i < 8; ++i) macc[ct][i] = 0.0f;

  for (int h = 0; h < H_; ++h) {
    float w[8];
#pragma unroll
    for (int i = 0; i < 8; ++i) {
      int e = et * 16 + i + hf * 8;
      w[i] = ex[e * H_ + h] / (denom[dstr[i] * H_ + h] + 1e-16f);
    }
#pragma unroll
    for (int ct = 0; ct < 4; ct += 2) {
      int cidx = h * 4 + ct;
      v8f c0 = {}, c1 = {};
      c0 = WMMA_BF16(a0, ldB_swz(We_s, cidx), c0);
      c1 = WMMA_BF16(a0, ldB_swz(We_s, cidx + 1), c1);
      c0 = WMMA_BF16(a1, ldB_swz(We_s, 32 + cidx), c0);
      c1 = WMMA_BF16(a1, ldB_swz(We_s, 32 + cidx + 1), c1);
      int col0 = cidx * 16 + cl, col1 = col0 + 16;
#pragma unroll
      for (int i = 0; i < 8; ++i)
        macc[ct][i] += (vmat[srcr[i] * HC_ + col0] + c0[i]) * w[i];
#pragma unroll
      for (int i = 0; i < 8; ++i)
        macc[ct + 1][i] += (vmat[srcr[i] * HC_ + col1] + c1[i]) * w[i];
    }
  }
#pragma unroll
  for (int ct = 0; ct < 4; ++ct)
#pragma unroll
    for (int i = 0; i < 8; ++i)
      atomicAdd(&out_acc[dstr[i] * 64 + ct * 16 + cl], macc[ct][i] * 0.125f);  // head mean
}

// ---------------------------------------------------------------------------
// Gated-residual MLP scores (weights pre-swizzled):
// sA = silu(x@w1+b1) [16x96], sB = silu(sA@w2+b2) [16x48], sg = sigmoid(...)
// ---------------------------------------------------------------------------
__device__ void gated_mlp(const float* u, const float* v,
                          const __bf16* w1, const float* b1,
                          const __bf16* w2, const float* b2,
                          const float* w3, const float* b3,
                          float* sA, float* sB, float* sg) {
  int lane = threadIdx.x, hf = lane >> 4, cl = lane & 15;
  for (int ct = 0; ct < 6; ct += 2) {        // 192 -> 96, KT=6 CT=6
    v8f c0 = {}, c1 = {};
    for (int kt = 0; kt < 6; ++kt) {
      v16bf a = ldA_gated(u, v, kt * 32);    // one A serves both column tiles
      c0 = WMMA_BF16(a, ldB_swz(w1, kt * 6 + ct), c0);
      c1 = WMMA_BF16(a, ldB_swz(w1, kt * 6 + ct + 1), c1);
    }
    int col0 = ct * 16 + cl, col1 = col0 + 16;
    float bb0 = b1[col0], bb1 = b1[col1];
#pragma unroll
    for (int i = 0; i < 8; ++i)
      sA[(i + hf * 8) * 96 + col0] = silu_f(c0[i] + bb0);
#pragma unroll
    for (int i = 0; i < 8; ++i)
      sA[(i + hf * 8) * 96 + col1] = silu_f(c1[i] + bb1);
  }
  __syncthreads();
  {                                          // 96 -> 48, KT=3 CT=3 (3 tiles in flight)
    v8f c0 = {}, c1 = {}, c2 = {};
    for (int kt = 0; kt < 3; ++kt) {
      v16bf a = ldA_f32(sA + kt * 32, 96);
      c0 = WMMA_BF16(a, ldB_swz(w2, kt * 3 + 0), c0);
      c1 = WMMA_BF16(a, ldB_swz(w2, kt * 3 + 1), c1);
      c2 = WMMA_BF16(a, ldB_swz(w2, kt * 3 + 2), c2);
    }
#pragma unroll
    for (int ct = 0; ct < 3; ++ct) {
      v8f c = (ct == 0) ? c0 : (ct == 1) ? c1 : c2;
      int col = ct * 16 + cl;
      float bb = b2[col];
#pragma unroll
      for (int i = 0; i < 8; ++i)
        sB[(i + hf * 8) * 48 + col] = silu_f(c[i] + bb);
    }
  }
  __syncthreads();
  if (lane < 16) {                           // 48 -> 1
    float s = b3[0];
    for (int j = 0; j < 48; ++j) s += sB[lane * 48 + j] * w3[j];
    sg[lane] = 1.0f / (1.0f + __expf(-s));
  }
  __syncthreads();
}

// ---------------------------------------------------------------------------
// Per-16-node epilogue: gated1 -> LN2 -> FFN -> gated2 -> out.
// ---------------------------------------------------------------------------
__global__ void __launch_bounds__(32)
k_epilogue(const float* __restrict__ h_ln, const float* __restrict__ out_acc,
           const float* __restrict__ skip,
           const float* ln2_g, const float* ln2_b,
           const __bf16* ga_w1, const float* ga_b1, const __bf16* ga_w2,
           const float* ga_b2, const float* ga_w3, const float* ga_b3,
           const __bf16* ff_w1, const float* ff_b1, const __bf16* ff_w2,
           const float* ff_b2,
           const __bf16* gf_w1, const float* gf_b1, const __bf16* gf_w2,
           const float* gf_b2, const float* gf_w3, const float* gf_b3,
           float* __restrict__ out) {
  __shared__ float su[16 * 64];   // u / h3
  __shared__ float sv[16 * 64];   // v_attn / ffn-out
  __shared__ float sA[16 * 96];   // mlp hidden / h2 / ffn hidden
  __shared__ float sB[16 * 48];
  __shared__ float sg[16];

  int nt = blockIdx.x, lane = threadIdx.x;
  int hf = lane >> 4, cl = lane & 15;
  int base = nt * 16 * 64;

  for (int i4 = lane; i4 < 16 * 16; i4 += 32) {  // float4 tile loads
    float4 a = ((const float4*)(h_ln + base))[i4];
    float4 b = ((const float4*)(out_acc + base))[i4];
    float4 s = ((const float4*)(skip + base))[i4];
    ((float4*)su)[i4] = a;
    float4 vv; vv.x = b.x + s.x; vv.y = b.y + s.y; vv.z = b.z + s.z; vv.w = b.w + s.w;
    ((float4*)sv)[i4] = vv;
  }
  __syncthreads();

  // gated residual 1
  gated_mlp(su, sv, ga_w1, ga_b1, ga_w2, ga_b2, ga_w3, ga_b3, sA, sB, sg);
  for (int idx = lane; idx < 16 * 64; idx += 32) {
    int r = idx >> 6;
    sA[idx] = sg[r] * su[idx] + (1.0f - sg[r]) * sv[idx];  // h2 (ld 64)
  }
  __syncthreads();

  // LN2 (one lane per row)
  if (lane < 16) {
    float mu = 0.0f;
    for (int c = 0; c < 64; ++c) mu += sA[lane * 64 + c];
    mu *= (1.0f / 64.0f);
    float var = 0.0f;
    for (int c = 0; c < 64; ++c) { float d = sA[lane * 64 + c] - mu; var += d * d; }
    float inv = rsqrtf(var * (1.0f / 64.0f) + 1e-5f);
    for (int c = 0; c < 64; ++c)
      su[lane * 64 + c] = (sA[lane * 64 + c] - mu) * inv * ln2_g[c] + ln2_b[c];  // h3
  }
  __syncthreads();

  // FFN hidden = silu(h3 @ ff_w1 + b1) -> sA (ld 64); KT=2 CT=4
  for (int ct = 0; ct < 4; ct += 2) {
    v8f c0 = {}, c1 = {};
    for (int kt = 0; kt < 2; ++kt) {
      v16bf a = ldA_f32(su + kt * 32, 64);
      c0 = WMMA_BF16(a, ldB_swz(ff_w1, kt * 4 + ct), c0);
      c1 = WMMA_BF16(a, ldB_swz(ff_w1, kt * 4 + ct + 1), c1);
    }
    int col0 = ct * 16 + cl, col1 = col0 + 16;
    float bb0 = ff_b1[col0], bb1 = ff_b1[col1];
#pragma unroll
    for (int i = 0; i < 8; ++i)
      sA[(i + hf * 8) * 64 + col0] = silu_f(c0[i] + bb0);
#pragma unroll
    for (int i = 0; i < 8; ++i)
      sA[(i + hf * 8) * 64 + col1] = silu_f(c1[i] + bb1);
  }
  __syncthreads();

  // FFN out = hidden @ ff_w2 + b2 -> sv
  for (int ct = 0; ct < 4; ct += 2) {
    v8f c0 = {}, c1 = {};
    for (int kt = 0; kt < 2; ++kt) {
      v16bf a = ldA_f32(sA + kt * 32, 64);
      c0 = WMMA_BF16(a, ldB_swz(ff_w2, kt * 4 + ct), c0);
      c1 = WMMA_BF16(a, ldB_swz(ff_w2, kt * 4 + ct + 1), c1);
    }
    int col0 = ct * 16 + cl, col1 = col0 + 16;
    float bb0 = ff_b2[col0], bb1 = ff_b2[col1];
#pragma unroll
    for (int i = 0; i < 8; ++i)
      sv[(i + hf * 8) * 64 + col0] = c0[i] + bb0;
#pragma unroll
    for (int i = 0; i < 8; ++i)
      sv[(i + hf * 8) * 64 + col1] = c1[i] + bb1;
  }
  __syncthreads();

  // gated residual 2 + final blend
  gated_mlp(su, sv, gf_w1, gf_b1, gf_w2, gf_b2, gf_w3, gf_b3, sA, sB, sg);
  for (int idx = lane; idx < 16 * 64; idx += 32) {
    int r = idx >> 6;
    out[base + idx] = sg[r] * su[idx] + (1.0f - sg[r]) * sv[idx];
  }
}

// ---------------------------------------------------------------------------
// Host launcher
// ---------------------------------------------------------------------------
static inline size_t align256(size_t x) { return (x + 255) & ~(size_t)255; }

extern "C" void kernel_launch(void* const* d_in, const int* in_sizes, int n_in,
                              void* d_out, int out_size, void* d_ws, size_t ws_size,
                              hipStream_t stream) {
  (void)in_sizes; (void)n_in; (void)out_size; (void)ws_size;

  const float* h        = (const float*)d_in[0];
  const int*   ei       = (const int*)d_in[1];
  const float* eattr    = (const float*)d_in[2];
  const float* ln1_g    = (const float*)d_in[3];
  const float* ln1_b    = (const float*)d_in[4];
  const float* ln2_g    = (const float*)d_in[5];
  const float* ln2_b    = (const float*)d_in[6];
  const float* Wq = (const float*)d_in[7];  const float* bq = (const float*)d_in[8];
  const float* Wk = (const float*)d_in[9];  const float* bk = (const float*)d_in[10];
  const float* Wv = (const float*)d_in[11]; const float* bv = (const float*)d_in[12];
  const float* We = (const float*)d_in[13];
  const float* Wskip = (const float*)d_in[14]; const float* bskip = (const float*)d_in[15];
  const float* ga_w1 = (const float*)d_in[16]; const float* ga_b1 = (const float*)d_in[17];
  const float* ga_w2 = (const float*)d_in[18]; const float* ga_b2 = (const float*)d_in[19];
  const float* ga_w3 = (const float*)d_in[20]; const float* ga_b3 = (const float*)d_in[21];
  const float* ff_w1 = (const float*)d_in[22]; const float* ff_b1 = (const float*)d_in[23];
  const float* ff_w2 = (const float*)d_in[24]; const float* ff_b2 = (const float*)d_in[25];
  const float* gf_w1 = (const float*)d_in[26]; const float* gf_b1 = (const float*)d_in[27];
  const float* gf_w2 = (const float*)d_in[28]; const float* gf_b2 = (const float*)d_in[29];
  const float* gf_w3 = (const float*)d_in[30]; const float* gf_b3 = (const float*)d_in[31];
  float* out = (float*)d_out;

  // workspace bump allocator
  char* ws = (char*)d_ws;
  size_t off = 0;
  auto alloc = [&](size_t bytes) { void* p = ws + off; off += align256(bytes); return p; };

  float*  h_ln     = (float*)alloc((size_t)N_ * 64 * 4);
  __bf16* h_bf     = (__bf16*)alloc((size_t)N_ * 64 * 2);
  float*  q        = (float*)alloc((size_t)N_ * HC_ * 4);
  float*  kbuf     = (float*)alloc((size_t)N_ * HC_ * 4);
  float*  vbuf     = (float*)alloc((size_t)N_ * HC_ * 4);
  float*  skip     = (float*)alloc((size_t)N_ * 64 * 4);
  __bf16* eattr_bf = (__bf16*)alloc((size_t)E_ * 64 * 2);
  float*  alphab   = (float*)alloc((size_t)E_ * H_ * 4);
  float*  amax     = (float*)alloc((size_t)N_ * H_ * 4);
  float*  denom    = (float*)alloc((size_t)N_ * H_ * 4);
  float*  out_acc  = (float*)alloc((size_t)N_ * 64 * 4);
  // swizzled bf16 weights (B-fragment order), same footprint as the raw weight
  __bf16* Wq_s    = (__bf16*)alloc((size_t)64 * HC_ * 2);
  __bf16* Wk_s    = (__bf16*)alloc((size_t)64 * HC_ * 2);
  __bf16* Wv_s    = (__bf16*)alloc((size_t)64 * HC_ * 2);
  __bf16* We_s    = (__bf16*)alloc((size_t)64 * HC_ * 2);
  __bf16* Ws_s    = (__bf16*)alloc((size_t)64 * 64 * 2);
  __bf16* ga_w1_s = (__bf16*)alloc((size_t)192 * 96 * 2);
  __bf16* ga_w2_s = (__bf16*)alloc((size_t)96 * 48 * 2);
  __bf16* ff_w1_s = (__bf16*)alloc((size_t)64 * 64 * 2);
  __bf16* ff_w2_s = (__bf16*)alloc((size_t)64 * 64 * 2);
  __bf16* gf_w1_s = (__bf16*)alloc((size_t)192 * 96 * 2);
  __bf16* gf_w2_s = (__bf16*)alloc((size_t)96 * 48 * 2);

  auto swz = [&](const float* s, __bf16* d, int KT, int CT, int ncols) {
    k_swzB<<<KT * CT, 32, 0, stream>>>(s, d, CT, ncols);
  };
  swz(Wq, Wq_s, 2, 32, HC_);   swz(Wk, Wk_s, 2, 32, HC_);
  swz(Wv, Wv_s, 2, 32, HC_);   swz(We, We_s, 2, 32, HC_);
  swz(Wskip, Ws_s, 2, 4, 64);
  swz(ga_w1, ga_w1_s, 6, 6, 96); swz(ga_w2, ga_w2_s, 3, 3, 48);
  swz(ff_w1, ff_w1_s, 2, 4, 64); swz(ff_w2, ff_w2_s, 2, 4, 64);
  swz(gf_w1, gf_w1_s, 6, 6, 96); swz(gf_w2, gf_w2_s, 3, 3, 48);

  k_f32_to_bf16_v4<<<((E_ * 64 / 4) + 255) / 256, 256, 0, stream>>>(eattr, eattr_bf, E_ * 64 / 4);

  k_fill<<<(N_ * H_ + 255) / 256, 256, 0, stream>>>(amax, -1e30f, N_ * H_);
  k_fill<<<(N_ * H_ + 255) / 256, 256, 0, stream>>>(denom, 0.0f, N_ * H_);
  k_fill<<<(N_ * 64 + 255) / 256, 256, 0, stream>>>(out_acc, 0.0f, N_ * 64);

  k_ln<<<N_, 32, 0, stream>>>(h, ln1_g, ln1_b, h_ln, h_bf);

  k_proj<<<dim3(4, NT_), 32, 0, stream>>>(h_bf, Wq_s, Wk_s, Wv_s, Ws_s,
                                          bq, bk, bv, bskip, q, kbuf, vbuf, skip);

  k_edge_alpha<<<ET_, 32, 0, stream>>>(eattr_bf, We_s, ei, q, kbuf, alphab, amax);

  k_softmax<<<(E_ * H_ + 255) / 256, 256, 0, stream>>>(ei, alphab, amax, denom);

  k_edge_msg<<<ET_, 32, 0, stream>>>(eattr_bf, We_s, ei, vbuf, alphab, denom, out_acc);

  k_epilogue<<<NT_, 32, 0, stream>>>(h_ln, out_acc, skip, ln2_g, ln2_b,
                                     ga_w1_s, ga_b1, ga_w2_s, ga_b2, ga_w3, ga_b3,
                                     ff_w1_s, ff_b1, ff_w2_s, ff_b2,
                                     gf_w1_s, gf_b1, gf_w2_s, gf_b2, gf_w3, gf_b3,
                                     out);
}
